// QuantizedLinear_65566970741152
// MI455X (gfx1250) — matrix operand order repaired and checked by reading, not compiled, not run on hardware
//
#include <hip/hip_runtime.h>

typedef __attribute__((ext_vector_type(16))) _Float16 v16h;
typedef __attribute__((ext_vector_type(8)))  _Float16 v8h;
typedef __attribute__((ext_vector_type(4)))  _Float16 v4h;
typedef __attribute__((ext_vector_type(8)))  float    v8f;

#define IN_F    4096
#define OUT_F   11008
#define GROUP   128
#define NGROUPS (IN_F / GROUP)

#define BM 128          // block tile M (tokens)
#define BN 128          // block tile N (out features)
#define BK 64           // K chunk staged in LDS (2 WMMA k-steps); group = 2 chunks
#define LDS_PAD 8       // halves -> row pitch 144 B (16B aligned, kills bank conflicts)
#define LDSK (BK + LDS_PAD)

#define THREADS 256     // 8 wave32 waves: 4 along M x 2 along N
#define WM 32           // wave tile M (2 x 16)
#define WN 64           // wave tile N (4 x 16)

__global__ __launch_bounds__(THREADS)
void qlinear_wmma_f16(const float* __restrict__ x,
                      const int*   __restrict__ qw,
                      const float* __restrict__ scales,
                      const float* __restrict__ zeros,
                      const float* __restrict__ bias,
                      float* __restrict__ out,
                      int Mtotal)
{
    __shared__ _Float16 Xs[BM][LDSK];   // x tile, f16, [m][k]
    __shared__ _Float16 Ws[BN][LDSK];   // dequantized weight tile, f16, [n][k]

    const int tid   = threadIdx.x;
    const int lane  = tid & 31;
    const int wave  = tid >> 5;
    const int waveM = wave & 3;    // 0..3
    const int waveN = wave >> 2;   // 0..1

    const int blockN0 = blockIdx.x * BN;
    const int blockM0 = blockIdx.y * BM;

    // WMMA lane decode (wave32):
    const int frow = lane & 15;    // A: row M, B: col N, D: col N
    const int hi   = lane >> 4;    // half-wave selector

    v8f acc[2][4];
    #pragma unroll
    for (int mi = 0; mi < 2; ++mi)
        #pragma unroll
        for (int ni = 0; ni < 4; ++ni)
            acc[mi][ni] = (v8f){0.f, 0.f, 0.f, 0.f, 0.f, 0.f, 0.f, 0.f};

    const int nchunks = IN_F / BK;   // 64
    for (int chunk = 0; chunk < nchunks; ++chunk) {
        const int k0    = chunk * BK;
        const int group = k0 / GROUP;          // quant group for this whole chunk

        // ---- prefetch next chunk's streams (global_prefetch_b8) ----
        if (chunk + 1 < nchunks) {
            const int pr = tid >> 1;           // 2 threads cover each of 128 rows
            __builtin_prefetch(qw + (size_t)(blockN0 + pr) * IN_F + k0 + BK, 0, 1);
            __builtin_prefetch(x  + (size_t)(blockM0 + pr) * IN_F + k0 + BK, 0, 1);
        }

        // ---- stage X tile: f32 -> f16 into LDS ----
        // BM*BK/4 = 2048 float4 slots, 8 per thread
        #pragma unroll
        for (int t = 0; t < (BM * BK / 4) / THREADS; ++t) {
            const int idx = t * THREADS + tid;
            const int m   = idx >> 4;          // BK/4 = 16 vec4 per row
            const int kv  = idx & 15;
            const float4 v = *(const float4*)(x + (size_t)(blockM0 + m) * IN_F + k0 + kv * 4);
            v4h h = { (_Float16)v.x, (_Float16)v.y, (_Float16)v.z, (_Float16)v.w };
            *(v4h*)(&Xs[m][kv * 4]) = h;
        }

        // ---- stage W tile: int4 codes -> dequant f16 into LDS ----
        #pragma unroll
        for (int t = 0; t < (BN * BK / 4) / THREADS; ++t) {
            const int idx = t * THREADS + tid;
            const int n   = idx >> 4;
            const int kv  = idx & 15;
            const int4 q  = *(const int4*)(qw + (size_t)(blockN0 + n) * IN_F + k0 + kv * 4);
            const float s = scales[(size_t)(blockN0 + n) * NGROUPS + group];
            const float z = zeros [(size_t)(blockN0 + n) * NGROUPS + group];
            v4h h = { (_Float16)(((float)q.x - z) * s),
                      (_Float16)(((float)q.y - z) * s),
                      (_Float16)(((float)q.z - z) * s),
                      (_Float16)(((float)q.w - z) * s) };
            *(v4h*)(&Ws[n][kv * 4]) = h;
        }

        __syncthreads();

        // ---- compute: BK/32 = 2 WMMA k-steps ----
        #pragma unroll
        for (int ks = 0; ks < BK / 32; ++ks) {
            const int kk = ks * 32;

            // A fragments (16x32 f16): lane holds M=frow,
            // K = {hi*8 .. hi*8+7} U {16+hi*8 .. 16+hi*8+7}
            v16h afrag[2];
            #pragma unroll
            for (int mi = 0; mi < 2; ++mi) {
                const _Float16* row = &Xs[waveM * WM + mi * 16 + frow][kk + hi * 8];
                v8h lo = *(const v8h*)(row);
                v8h hh = *(const v8h*)(row + 16);
                afrag[mi] = __builtin_shufflevector(lo, hh,
                    0, 1, 2, 3, 4, 5, 6, 7, 8, 9, 10, 11, 12, 13, 14, 15);
            }

            // B fragments (32x16 f16): lane holds N=frow, K = {hi*8 .. hi*8+7} U {16+hi*8 .. 16+hi*8+7}
            // B[k][n] = W^T -> contiguous k run inside Ws[n][...]
            v16h bfrag[4];
            #pragma unroll
            for (int ni = 0; ni < 4; ++ni) {
                const _Float16* row = &Ws[waveN * WN + ni * 16 + frow][kk + hi * 8];
                v8h lo = *(const v8h*)(row);
                v8h hh = *(const v8h*)(row + 16);
                bfrag[ni] = __builtin_shufflevector(lo, hh,
                    0, 1, 2, 3, 4, 5, 6, 7, 8, 9, 10, 11, 12, 13, 14, 15);
            }

            #pragma unroll
            for (int mi = 0; mi < 2; ++mi)
                #pragma unroll
                for (int ni = 0; ni < 4; ++ni)
                    acc[mi][ni] = __builtin_amdgcn_wmma_f32_16x16x32_f16(
                        /*neg_a=*/false, afrag[mi],
                        /*neg_b=*/false, bfrag[ni],
                        /*c_mod=*/(short)0, acc[mi][ni],
                        /*reuse_a=*/false, /*reuse_b=*/false);
        }

        __syncthreads();
    }

    // ---- epilogue: D VGPR r <-> (M = 8*hi + r, N = frow); add bias, store f32 ----
    const int colBase = blockN0 + waveN * WN;
    const int rowBase = blockM0 + waveM * WM;
    #pragma unroll
    for (int ni = 0; ni < 4; ++ni) {
        const int col = colBase + ni * 16 + frow;
        const float bv = bias[col];
        #pragma unroll
        for (int mi = 0; mi < 2; ++mi) {
            const int rbase = rowBase + mi * 16 + hi * 8;
            #pragma unroll
            for (int r = 0; r < 8; ++r) {
                const int row = rbase + r;
                out[(size_t)row * OUT_F + col] = acc[mi][ni][r] + bv;
            }
        }
    }
}

extern "C" void kernel_launch(void* const* d_in, const int* in_sizes, int n_in,
                              void* d_out, int out_size, void* d_ws, size_t ws_size,
                              hipStream_t stream) {
    const float* x      = (const float*)d_in[0];
    const int*   qw     = (const int*)  d_in[1];
    const float* scales = (const float*)d_in[2];
    const float* zeros  = (const float*)d_in[3];
    const float* bias   = (const float*)d_in[4];
    float*       out    = (float*)d_out;

    const int Mtotal = in_sizes[0] / IN_F;       // 4*2048 = 8192, divisible by BM
    dim3 grid(OUT_F / BN, Mtotal / BM);          // (86, 64)
    qlinear_wmma_f16<<<grid, THREADS, 0, stream>>>(x, qw, scales, zeros, bias, out, Mtotal);
}
